// MultiHeadSelfAttention_64433099374809
// MI455X (gfx1250) — compile-verified, matrix operand order checked by reading, not run on hardware
//
#include <hip/hip_runtime.h>
#include <hip/hip_bf16.h>

// MI455X MHA forward. bf16 WMMA (16x16x32) everywhere, fp32 accumulate.
// Pipeline: cvt -> 3x GEMM (QKV) -> RoPE -> V-transpose -> flash-attn -> GEMM (WO).

#define BB 4
#define SS 2048
#define DD 1024
#define HH 16
#define DKK 64

typedef unsigned short u16;
typedef __attribute__((ext_vector_type(16))) __bf16 v16bf;
typedef __attribute__((ext_vector_type(8)))  float  v8f;
typedef __attribute__((ext_vector_type(4)))  unsigned int u32x4;
typedef __attribute__((ext_vector_type(4)))  unsigned short u16x4;
typedef __attribute__((ext_vector_type(4)))  float f32x4;

#if __has_builtin(__builtin_amdgcn_global_load_async_to_lds_b128)
#define HAVE_ASYNC_LDS 1
typedef int i32x4g __attribute__((vector_size(16)));
typedef __attribute__((address_space(1))) i32x4g* gas_i32x4;
typedef __attribute__((address_space(3))) i32x4g* las_i32x4;
#endif

static __device__ __forceinline__ u16 f2bf(float f) {
  union { float f; unsigned u; } x; x.f = f;
  unsigned r = x.u + 0x7FFFu + ((x.u >> 16) & 1u);   // RNE
  return (u16)(r >> 16);
}
static __device__ __forceinline__ float bf2f(u16 h) {
  union { unsigned u; float f; } x; x.u = ((unsigned)h) << 16;
  return x.f;
}

union FragU { u32x4 u[2]; v16bf v; };

// Load a 16-half fragment: per-lane K pattern {kb..kb+7, kb+16..kb+23},
// caller passes p = row_base + k0 + kb (kb = 8*(lane>=16)). Two b128 loads.
static __device__ __forceinline__ v16bf load_frag(const u16* p) {
  FragU t;
  t.u[0] = *(const u32x4*)(p);
  t.u[1] = *(const u32x4*)(p + 16);
  return t.v;
}
static __device__ __forceinline__ v16bf load_frag_lds(const u16* p) {
  FragU t;
  t.u[0] = *(const u32x4*)(p);
  t.u[1] = *(const u32x4*)(p + 16);
  return t.v;
}

static __device__ __forceinline__ v8f wmma_bf16(v16bf a, v16bf b, v8f c) {
  return __builtin_amdgcn_wmma_f32_16x16x32_bf16(false, a, false, b, (short)0, c, false, false);
}

// ---------------- fp32 -> bf16 convert (vector x4) ----------------
__global__ __launch_bounds__(256)
void cvt_f32_bf16_kernel(const float* __restrict__ in, u16* __restrict__ out, int n4) {
  int i = blockIdx.x * 256 + threadIdx.x;
  if (i >= n4) return;
  f32x4 v = *(const f32x4*)(in + 4 * (size_t)i);
  u16x4 o;
  o[0] = f2bf(v[0]); o[1] = f2bf(v[1]); o[2] = f2bf(v[2]); o[3] = f2bf(v[3]);
  *(u16x4*)(out + 4 * (size_t)i) = o;
}

// ---------------- GEMM: C[M,N] = A[M,K] * W[N,K]^T (bf16 in, f32 acc) ----
// Block: 256 thr = 8 waves. Block tile 128M x 64N, wave tile 16M x 64N.
// W tile (64x32 bf16 = 4KB) staged in LDS (async-to-LDS when available).
__global__ __launch_bounds__(256)
void gemm_bf16_kernel(const u16* __restrict__ A, const u16* __restrict__ W,
                      float* __restrict__ Cf, u16* __restrict__ Cb,
                      int M, int N, int K) {
  __shared__ u16 sB[64 * 32];
  const int t = threadIdx.x;
  const int wave = t >> 5, lane = t & 31;
  const int lq = lane & 15, hi = lane >> 4;
  const int kb = hi * 8;
  const int m0 = blockIdx.x * 128 + wave * 16;
  const int n0 = blockIdx.y * 64;

  v8f acc[4] = {v8f{}, v8f{}, v8f{}, v8f{}};

  const u16* arow = A + (size_t)(m0 + lq) * K;
  const int srow = t >> 2, sc = (t & 3) * 8;             // 256 thr x 16B = 4KB tile
  const u16* wsrc = W + (size_t)(n0 + srow) * K + sc;
  u16* sdst = &sB[srow * 32 + sc];

  for (int kk = 0; kk < K; kk += 32) {
    __syncthreads();                                     // previous reads done
#if HAVE_ASYNC_LDS
    __builtin_amdgcn_global_load_async_to_lds_b128(
        (gas_i32x4)(wsrc + kk), (las_i32x4)sdst, 0, 0);
#if __has_builtin(__builtin_amdgcn_s_wait_asynccnt)
    __builtin_amdgcn_s_wait_asynccnt(0);
#else
    asm volatile("s_wait_asynccnt 0" ::: "memory");
#endif
#else
    *(u32x4*)sdst = *(const u32x4*)(wsrc + kk);
#endif
    __syncthreads();                                     // tile visible

    v16bf af = load_frag(arow + kk + kb);
#pragma unroll
    for (int j = 0; j < 4; ++j) {
      v16bf bf = load_frag_lds(&sB[(j * 16 + lq) * 32 + kb]);
      acc[j] = wmma_bf16(af, bf, acc[j]);
    }
  }

#pragma unroll
  for (int j = 0; j < 4; ++j) {
#pragma unroll
    for (int r = 0; r < 8; ++r) {
      int row = m0 + r + 8 * hi;                         // C layout: m = r + 8*(lane>=16)
      int col = n0 + j * 16 + lq;                        // n = lane%16
      if (Cf) Cf[(size_t)row * N + col] = acc[j][r];
      else    Cb[(size_t)row * N + col] = f2bf(acc[j][r]);
    }
  }
}

// ---------------- RoPE on Q and K (bf16 in/out, fp32 math) ----------------
__global__ __launch_bounds__(256)
void rope_kernel(const u16* __restrict__ Qi, const u16* __restrict__ Ki,
                 const int* __restrict__ pos,
                 u16* __restrict__ Qo, u16* __restrict__ Ko) {
  int i = blockIdx.x * 256 + threadIdx.x;                // over B*S*(D/2)
  if (i >= BB * SS * (DD / 2)) return;
  int dh = i & (DD / 2 - 1);
  int bs = i >> 9;                                        // D/2 = 512
  int h = dh >> 5, j = dh & 31;
  float p = (float)pos[bs];
  float ang = p * __expf((float)j * (-0.2878231366f));    // -ln(10000)/32
  float sn, cs;
  __sincosf(ang, &sn, &cs);
  size_t off = (size_t)bs * DD + h * DKK + 2 * j;

  unsigned qv = *(const unsigned*)(Qi + off);
  float q1 = bf2f((u16)(qv & 0xffff)), q2 = bf2f((u16)(qv >> 16));
  unsigned qo = (unsigned)f2bf(q1 * cs - q2 * sn) | ((unsigned)f2bf(q1 * sn + q2 * cs) << 16);
  *(unsigned*)(Qo + off) = qo;

  unsigned kv = *(const unsigned*)(Ki + off);
  float k1 = bf2f((u16)(kv & 0xffff)), k2 = bf2f((u16)(kv >> 16));
  unsigned ko = (unsigned)f2bf(k1 * cs - k2 * sn) | ((unsigned)f2bf(k1 * sn + k2 * cs) << 16);
  *(unsigned*)(Ko + off) = ko;
}

// ---------------- V transpose: Vt[b,h,d,s] = Vb[b,s,h*64+d] ----------------
__global__ __launch_bounds__(256)
void vtrans_kernel(const u16* __restrict__ Vb, u16* __restrict__ Vt) {
  int i = blockIdx.x * 256 + threadIdx.x;                // over B*H*DK*S
  if (i >= BB * HH * DKK * SS) return;
  int s = i & (SS - 1);
  int r1 = i >> 11;
  int d = r1 & (DKK - 1);
  int bh = r1 >> 6;
  int b = bh >> 4, h = bh & (HH - 1);
  Vt[i] = Vb[((size_t)(b * SS + s)) * DD + h * DKK + d];
}

// ---------------- Flash attention: 1 wave per 16-row q tile ----------------
// scores: S = Q K^T (2 subtiles x 2 K-halves = 4 wmma / 32 keys)
// softmax: online, row stats per C-fragment row (shfl_xor over 16-lane half)
// P*V: P staged via LDS (C-layout -> A-layout), 4 wmma over DK=64.
__global__ __launch_bounds__(32)
void attn_kernel(const u16* __restrict__ Qb, const u16* __restrict__ Kb,
                 const u16* __restrict__ Vt, u16* __restrict__ Ob) {
  __shared__ u16 Pl[16 * 32];
  const int lane = threadIdx.x;
  const int lq = lane & 15, hi = lane >> 4;
  const int kb = hi * 8;
  const int qb = blockIdx.x * 16;
  const int bh = blockIdx.y;
  const int b = bh >> 4, h = bh & (HH - 1);

  const u16* qrow = Qb + ((size_t)(b * SS + qb + lq)) * DD + h * DKK;
  v16bf qf0 = load_frag(qrow + kb);
  v16bf qf1 = load_frag(qrow + 32 + kb);

  v8f o[4] = {v8f{}, v8f{}, v8f{}, v8f{}};
  float mi[8], li[8];
#pragma unroll
  for (int r = 0; r < 8; ++r) { mi[r] = -1.0e30f; li[r] = 0.0f; }

  const int qend = qb + 15;
  for (int kk = 0; kk <= qend; kk += 32) {
    v8f s0 = v8f{}, s1 = v8f{};
    {
      const u16* k0 = Kb + ((size_t)(b * SS + kk + lq)) * DD + h * DKK;
      s0 = wmma_bf16(qf0, load_frag(k0 + kb), s0);
      s0 = wmma_bf16(qf1, load_frag(k0 + 32 + kb), s0);
      const u16* k1 = Kb + ((size_t)(b * SS + kk + 16 + lq)) * DD + h * DKK;
      s1 = wmma_bf16(qf0, load_frag(k1 + kb), s1);
      s1 = wmma_bf16(qf1, load_frag(k1 + 32 + kb), s1);
    }

    float p0[8], p1[8], fac[8];
#pragma unroll
    for (int r = 0; r < 8; ++r) {
      const int q = qb + r + 8 * hi;
      float a = s0[r] * 0.125f;                          // 1/sqrt(64)
      float c = s1[r] * 0.125f;
      if (kk + lq > q)       a = -1.0e30f;               // causal mask
      if (kk + 16 + lq > q)  c = -1.0e30f;
      float mx = fmaxf(a, c);
      mx = fmaxf(mx, __shfl_xor(mx, 1));
      mx = fmaxf(mx, __shfl_xor(mx, 2));
      mx = fmaxf(mx, __shfl_xor(mx, 4));
      mx = fmaxf(mx, __shfl_xor(mx, 8));
      float mn = fmaxf(mi[r], mx);
      a = __expf(a - mn);
      c = __expf(c - mn);
      float rs = a + c;
      rs += __shfl_xor(rs, 1);
      rs += __shfl_xor(rs, 2);
      rs += __shfl_xor(rs, 4);
      rs += __shfl_xor(rs, 8);
      fac[r] = __expf(mi[r] - mn);
      li[r] = li[r] * fac[r] + rs;
      mi[r] = mn;
      p0[r] = a; p1[r] = c;
    }
#pragma unroll
    for (int j = 0; j < 4; ++j)
#pragma unroll
      for (int r = 0; r < 8; ++r) o[j][r] *= fac[r];

    // C-layout -> LDS (row m = r + 8*hi, cols lq and 16+lq)
#pragma unroll
    for (int r = 0; r < 8; ++r) {
      int m = r + 8 * hi;
      Pl[m * 32 + lq]      = f2bf(p0[r]);
      Pl[m * 32 + 16 + lq] = f2bf(p1[r]);
    }
    __builtin_amdgcn_wave_barrier();                     // LDS in-order per wave

    v16bf pf = load_frag_lds(&Pl[lq * 32 + kb]);         // A-layout read
#pragma unroll
    for (int j = 0; j < 4; ++j) {
      const u16* vp = Vt + (((size_t)(b * HH + h) * DKK + j * 16 + lq)) * SS + kk;
      o[j] = wmma_bf16(pf, load_frag(vp + kb), o[j]);
    }
    __builtin_amdgcn_wave_barrier();
  }

  float inv[8];
#pragma unroll
  for (int r = 0; r < 8; ++r) inv[r] = 1.0f / li[r];
#pragma unroll
  for (int j = 0; j < 4; ++j)
#pragma unroll
    for (int r = 0; r < 8; ++r) {
      int q = qb + r + 8 * hi;
      Ob[((size_t)(b * SS + q)) * DD + h * DKK + j * 16 + lq] = f2bf(o[j][r] * inv[r]);
    }
}

// ---------------- host ----------------
extern "C" void kernel_launch(void* const* d_in, const int* in_sizes, int n_in,
                              void* d_out, int out_size, void* d_ws, size_t ws_size,
                              hipStream_t stream) {
  const float* x  = (const float*)d_in[0];
  const int* pos  = (const int*)d_in[1];
  const float* wq = (const float*)d_in[2];
  const float* wk = (const float*)d_in[3];
  const float* wv = (const float*)d_in[4];
  const float* wo = (const float*)d_in[5];

  const size_t NX = (size_t)BB * SS * DD;      // 8388608
  const size_t NW = (size_t)DD * DD;           // 1048576

  char* w = (char*)d_ws;
  u16* xb  = (u16*)w;            w += NX * 2;
  u16* wqb = (u16*)w;            w += NW * 2;
  u16* wkb = (u16*)w;            w += NW * 2;
  u16* wvb = (u16*)w;            w += NW * 2;
  u16* wob = (u16*)w;            w += NW * 2;
  u16* Qt  = (u16*)w;            w += NX * 2;
  u16* Kt  = (u16*)w;            w += NX * 2;
  u16* Vb  = (u16*)w;            w += NX * 2;
  u16* Qr  = (u16*)w;            w += NX * 2;
  u16* Kr  = (u16*)w;            w += NX * 2;
  u16* Vt  = (u16*)w;            w += NX * 2;
  u16* Ob  = (u16*)w;            w += NX * 2;

  // 1) convert to bf16
  cvt_f32_bf16_kernel<<<(int)(NX / 4 / 256), 256, 0, stream>>>(x,  xb,  (int)(NX / 4));
  cvt_f32_bf16_kernel<<<(int)(NW / 4 / 256), 256, 0, stream>>>(wq, wqb, (int)(NW / 4));
  cvt_f32_bf16_kernel<<<(int)(NW / 4 / 256), 256, 0, stream>>>(wk, wkb, (int)(NW / 4));
  cvt_f32_bf16_kernel<<<(int)(NW / 4 / 256), 256, 0, stream>>>(wv, wvb, (int)(NW / 4));
  cvt_f32_bf16_kernel<<<(int)(NW / 4 / 256), 256, 0, stream>>>(wo, wob, (int)(NW / 4));

  // 2) QKV projections
  const int M = BB * SS;
  dim3 gg(M / 128, DD / 64);
  gemm_bf16_kernel<<<gg, 256, 0, stream>>>(xb, wqb, nullptr, Qt, M, DD, DD);
  gemm_bf16_kernel<<<gg, 256, 0, stream>>>(xb, wkb, nullptr, Kt, M, DD, DD);
  gemm_bf16_kernel<<<gg, 256, 0, stream>>>(xb, wvb, nullptr, Vb, M, DD, DD);

  // 3) RoPE on Q,K
  rope_kernel<<<(BB * SS * (DD / 2)) / 256, 256, 0, stream>>>(Qt, Kt, pos, Qr, Kr);

  // 4) V transpose to [B,H,DK,S]
  vtrans_kernel<<<(int)(NX / 256), 256, 0, stream>>>(Vb, Vt);

  // 5) causal flash attention
  dim3 ga(SS / 16, BB * HH);
  attn_kernel<<<ga, 32, 0, stream>>>(Qr, Kr, Vt, Ob);

  // 6) output projection -> fp32 d_out
  gemm_bf16_kernel<<<gg, 256, 0, stream>>>(Ob, wob, (float*)d_out, nullptr, M, DD, DD);
}
